// Bn_bin_conv_pool_block_48309792145688
// MI455X (gfx1250) — matrix operand order repaired and checked by reading, not compiled, not run on hardware
//
#include <hip/hip_runtime.h>
#include <hip/hip_bf16.h>

typedef _Float16 v8h  __attribute__((ext_vector_type(8)));
typedef _Float16 v16h __attribute__((ext_vector_type(16)));
typedef float    v8f  __attribute__((ext_vector_type(8)));

#define N_BATCH 64
#define CIN     64
#define LLEN    8192
#define COUT    128
#define KW      7
#define KKTOT   448      // KW*CIN, ordered kk = k*64 + ci
#define LB      128      // l-positions per workgroup tile
#define HALO    3
#define XROWS   (LB + 2*HALO)   // 134
#define XSTR    72              // padded row stride (f16), mult of 8, bank-friendly
#define WSTR    456             // padded W row stride (f16), mult of 8, bank-friendly
#define YSTR    132             // f32 epilogue staging stride
#define SMEM_BYTES (COUT*WSTR*2 + XROWS*XSTR*2)   // 116736 + 19296 = 136032

union V16U { v16h v; v8h h[2]; };

// ---------------- Kernel 1: per-channel batch stats -> binarize threshold ----
__global__ __launch_bounds__(256)
void stats_kernel(const float* __restrict__ I, const float* __restrict__ gamma,
                  const float* __restrict__ beta, float* __restrict__ thr,
                  float* __restrict__ sgn)
{
    const int ci  = blockIdx.x;
    const int tid = threadIdx.x;
    double s = 0.0, s2 = 0.0;
    for (int idx = tid; idx < N_BATCH * LLEN; idx += 256) {
        int n = idx >> 13;            // idx / 8192
        int l = idx & (LLEN - 1);
        float x = I[(((size_t)n * CIN + ci) << 13) + l];
        s  += (double)x;
        s2 += (double)x * (double)x;
    }
    __shared__ double rs[256], rq[256];
    rs[tid] = s; rq[tid] = s2;
    __syncthreads();
    for (int off = 128; off > 0; off >>= 1) {
        if (tid < off) { rs[tid] += rs[tid + off]; rq[tid] += rq[tid + off]; }
        __syncthreads();
    }
    if (tid == 0) {
        const double cnt = (double)N_BATCH * (double)LLEN;
        double mean = rs[0] / cnt;
        double var  = rq[0] / cnt - mean * mean;
        float invstd = rsqrtf((float)var + 1e-5f);
        float a = gamma[ci] * invstd;                 // > 0 for reference inputs
        thr[ci] = (float)mean - beta[ci] / a;
        sgn[ci] = (a >= 0.f) ? 1.f : -1.f;
    }
}

// ---------------- Kernel 2: repack W f32 [co][ci][k] -> f16 [co][k*64+ci] ----
__global__ __launch_bounds__(256)
void wpack_kernel(const float* __restrict__ W, _Float16* __restrict__ Wh)
{
    int idx = blockIdx.x * 256 + threadIdx.x;
    if (idx >= COUT * CIN * KW) return;
    int k  = idx % KW;
    int ci = (idx / KW) % CIN;
    int co = idx / (CIN * KW);
    Wh[co * KKTOT + k * CIN + ci] = (_Float16)W[idx];
}

// ---------------- Kernel 3: binarize-on-load + WMMA implicit-GEMM conv + pool
__global__ __launch_bounds__(256)
void conv_wmma_kernel(const float* __restrict__ I, const float* __restrict__ thr,
                      const float* __restrict__ sgn, const _Float16* __restrict__ Wh,
                      const float* __restrict__ alpha, float* __restrict__ out)
{
    extern __shared__ char smem[];
    _Float16* Wl = (_Float16*)smem;                        // [COUT][WSTR]
    _Float16* Xl = (_Float16*)(smem + COUT * WSTR * 2);    // [XROWS][XSTR]
    float*    Ys = (float*)smem;                           // epilogue alias [COUT][YSTR]

    const int lBlock = blockIdx.x * LB;
    const int n      = blockIdx.y;
    const int tid    = threadIdx.x;

    // --- stage W tile (all of W): 128 rows x 448 f16 as 16B chunks, coalesced
    for (int c = tid; c < COUT * (KKTOT / 8); c += 256) {
        int row = c / (KKTOT / 8);
        int chk = c % (KKTOT / 8);
        v8h v = *(const v8h*)(Wh + row * KKTOT + chk * 8);
        *(v8h*)(Wl + row * WSTR + chk * 8) = v;
    }

    // --- stage x tile: binarize on the fly; zero outside [0,L) (conv zero-pad)
    for (int e = tid; e < CIN * XROWS; e += 256) {
        int ci = e / XROWS;
        int li = e % XROWS;
        int gl = lBlock - HALO + li;
        _Float16 v = (_Float16)0.f;
        if (gl >= 0 && gl < LLEN) {
            float x = I[(((size_t)n * CIN + ci) << 13) + gl];
            float d = (x - thr[ci]) * sgn[ci];
            v = (d >= 0.f) ? (_Float16)1.f : (_Float16)(-1.f);
        }
        Xl[li * XSTR + ci] = v;
    }
    __syncthreads();

    const int wave   = tid >> 5;
    const int lane   = tid & 31;
    const int lanelo = lane & 15;
    const int hi     = lane >> 4;
    const int coBase = (wave >> 1) * 32;   // 4 co-groups of 32
    const int lBase  = (wave & 1) * 64;    // 2 l-groups of 64

    v8f acc[2][4];
#pragma unroll
    for (int ct = 0; ct < 2; ++ct)
#pragma unroll
        for (int lt = 0; lt < 4; ++lt) {
            v8f z = {0.f, 0.f, 0.f, 0.f, 0.f, 0.f, 0.f, 0.f};
            acc[ct][lt] = z;
        }

    // --- GEMM K loop: 14 steps of 32 over kk = k*64 + ci
    for (int ks = 0; ks < 14; ++ks) {
        const int k      = ks >> 1;                       // conv tap
        const int ciBase = (ks & 1) * 32;                 // ci half of this step

        // A fragments: 16-bit A 16x32 layout => runs K = ks*32 + hi*8 (+16)
        V16U af[2];
#pragma unroll
        for (int ct = 0; ct < 2; ++ct) {
            const _Float16* base = Wl + (coBase + ct * 16 + lanelo) * WSTR + ks * 32;
            af[ct].h[0] = *(const v8h*)(base + hi * 8);
            af[ct].h[1] = *(const v8h*)(base + 16 + hi * 8);
        }
        // B fragments: same K order as A => K = hi*8 (+16)
        V16U bf[4];
#pragma unroll
        for (int lt = 0; lt < 4; ++lt) {
            const _Float16* base = Xl + (lBase + lt * 16 + lanelo + k) * XSTR + ciBase;
            bf[lt].h[0] = *(const v8h*)(base + hi * 8);
            bf[lt].h[1] = *(const v8h*)(base + 16 + hi * 8);
        }
#pragma unroll
        for (int ct = 0; ct < 2; ++ct)
#pragma unroll
            for (int lt = 0; lt < 4; ++lt) {
                acc[ct][lt] = __builtin_amdgcn_wmma_f32_16x16x32_f16(
                    false, af[ct].v, false, bf[lt].v,
                    (short)0, acc[ct][lt], false, false);
            }
    }

    // --- epilogue: stage f32 y into LDS (alias over W region), pool, scale
    __syncthreads();
#pragma unroll
    for (int ct = 0; ct < 2; ++ct)
#pragma unroll
        for (int lt = 0; lt < 4; ++lt) {
            int co = coBase + ct * 16 + 8 * hi;   // rows co+r, r=0..7
            int l  = lBase + lt * 16 + lanelo;
#pragma unroll
            for (int r = 0; r < 8; ++r)
                Ys[(co + r) * YSTR + l] = acc[ct][lt][r];
        }
    __syncthreads();

    for (int e = tid; e < COUT * (LB / 2); e += 256) {
        int co = e >> 6;          // / 64
        int lp = e & 63;
        float m = fmaxf(Ys[co * YSTR + 2 * lp], Ys[co * YSTR + 2 * lp + 1]);
        out[(((size_t)n * COUT + co) << 12) + blockIdx.x * (LB / 2) + lp] = m * alpha[co];
    }
}

// ---------------- launcher ----------------
extern "C" void kernel_launch(void* const* d_in, const int* in_sizes, int n_in,
                              void* d_out, int out_size, void* d_ws, size_t ws_size,
                              hipStream_t stream)
{
    const float* I     = (const float*)d_in[0];
    const float* gamma = (const float*)d_in[1];
    const float* beta  = (const float*)d_in[2];
    const float* W     = (const float*)d_in[3];
    const float* alpha = (const float*)d_in[4];
    float* out = (float*)d_out;

    float*     thr = (float*)d_ws;
    float*     sgn = thr + CIN;
    _Float16*  Wh  = (_Float16*)((char*)d_ws + 512);

    stats_kernel<<<CIN, 256, 0, stream>>>(I, gamma, beta, thr, sgn);

    int wtot = COUT * CIN * KW;
    wpack_kernel<<<(wtot + 255) / 256, 256, 0, stream>>>(W, Wh);

    dim3 grid(LLEN / LB, N_BATCH);   // 64 x 64
    conv_wmma_kernel<<<grid, 256, SMEM_BYTES, stream>>>(I, thr, sgn, Wh, alpha, out);
}
